// ModulatedConvolutionLayer_88192858456284
// MI455X (gfx1250) — matrix operand order repaired and checked by reading, not compiled, not run on hardware
//
#include <hip/hip_runtime.h>
#include <hip/hip_bf16.h>

typedef __attribute__((ext_vector_type(16))) _Float16 v16h;
typedef __attribute__((ext_vector_type(8)))  _Float16 v8h;
typedef __attribute__((ext_vector_type(8)))  float    v8f;

#define BB 8
#define CI 256
#define CO 256
#define SS 512
#define HH 128
#define WW 128
#define HO 64
#define WO 64

// ---------------------------------------------------------------------------
// Kernel 1: 5x5 binomial depthwise blur (pad 2) + per-plane sum of squares.
// One block per (b,c) plane. Writes blurred plane as f16 (un-normalized;
// the second-moment scale is folded into the weights in k_wprep).
// ---------------------------------------------------------------------------
__global__ __launch_bounds__(256) void k_blur(const float* __restrict__ x,
                                              _Float16* __restrict__ xb,
                                              float* __restrict__ sumsq) {
  __shared__ float tile[36][WW];  // 18KB: 32-row band + 2-halo each side
  const int plane = blockIdx.x;   // b*256 + c
  const float* xp = x + (size_t)plane * HH * WW;
  _Float16* xo = xb + (size_t)plane * HH * WW;
  const int tid = threadIdx.x;
  const float kw0 = 1.0f / 16.0f, kw1 = 4.0f / 16.0f, kw2 = 6.0f / 16.0f;
  const float kw[5] = {kw0, kw1, kw2, kw1, kw0};
  float ss = 0.0f;
  for (int band = 0; band < 4; ++band) {
    const int ybase = band * 32;
    __syncthreads();
    for (int idx = tid; idx < 36 * WW; idx += 256) {
      int r = idx >> 7, c = idx & 127;
      int gy = ybase - 2 + r;
      tile[r][c] = ((unsigned)gy < (unsigned)HH) ? xp[gy * WW + c] : 0.0f;
    }
    __syncthreads();
    for (int idx = tid; idx < 32 * WW; idx += 256) {
      int ry = idx >> 7, cx = idx & 127;
      float acc = 0.0f;
#pragma unroll
      for (int dy = 0; dy < 5; ++dy) {
        float h = 0.0f;
#pragma unroll
        for (int dx = 0; dx < 5; ++dx) {
          int xx = cx - 2 + dx;
          float v = ((unsigned)xx < (unsigned)WW) ? tile[ry + dy][xx] : 0.0f;
          h += kw[dx] * v;
        }
        acc += kw[dy] * h;
      }
      ss += acc * acc;
      xo[(ybase + ry) * WW + cx] = (_Float16)acc;
    }
  }
  // wave reduce, then one atomic per wave
#pragma unroll
  for (int off = 16; off > 0; off >>= 1) ss += __shfl_down(ss, off, 32);
  if ((tid & 31) == 0) atomicAdd(&sumsq[plane], ss);
}

// ---------------------------------------------------------------------------
// Kernel 2: style[b,i] = dot(s[b,:], style_w[i,:]) + style_b[i]   (8x256x512)
// ---------------------------------------------------------------------------
__global__ __launch_bounds__(256) void k_style(const float* __restrict__ s,
                                               const float* __restrict__ sw,
                                               const float* __restrict__ sb,
                                               float* __restrict__ style) {
  int idx = blockIdx.x * 256 + threadIdx.x;
  if (idx >= BB * CI) return;
  int b = idx >> 8, i = idx & 255;
  const float* sp = s + (size_t)b * SS;
  const float* wp = sw + (size_t)i * SS;
  float acc = 0.0f;
  for (int k = 0; k < SS; ++k) acc += sp[k] * wp[k];
  style[idx] = acc + sb[i];
}

// ---------------------------------------------------------------------------
// Kernel 3: modulated+demodulated weights, with the second-moment norm scale
// folded in on the input-channel axis. One block per (b,o), thread == i.
// Output layout weffL[b][tap][o][i] (A-fragment friendly: k==i contiguous).
// ---------------------------------------------------------------------------
__global__ __launch_bounds__(256) void k_wprep(const float* __restrict__ w,
                                               const float* __restrict__ style,
                                               const float* __restrict__ sumsq,
                                               const float* __restrict__ g,
                                               _Float16* __restrict__ weffL) {
  __shared__ float red[8];
  const int b = blockIdx.x >> 8;
  const int o = blockIdx.x & 255;
  const int i = threadIdx.x;
  const float scale = 0.0208333333333f;  // 1/sqrt(256*9) = 1/48
  const float st = style[b * CI + i];
  const float* wp = w + ((size_t)o * CI + i) * 9;
  float wm[9];
  float ss = 0.0f;
#pragma unroll
  for (int t = 0; t < 9; ++t) {
    wm[t] = scale * wp[t] * st;
    ss += wm[t] * wm[t];
  }
#pragma unroll
  for (int off = 16; off > 0; off >>= 1) ss += __shfl_down(ss, off, 32);
  if ((i & 31) == 0) red[i >> 5] = ss;
  __syncthreads();
  if (i == 0) {
    float tt = 0.0f;
    for (int r = 0; r < 8; ++r) tt += red[r];
    red[0] = tt;
  }
  __syncthreads();
  const float d = rsqrtf(red[0] + 1e-8f);                 // demodulation
  const float ms = sumsq[b * CI + i] * (1.0f / 16384.0f); // mean of squares
  const float rs = rsqrtf(ms + 1e-8f) * g[i];             // norm scale * gain
  const float f = d * rs;
#pragma unroll
  for (int t = 0; t < 9; ++t)
    weffL[(((size_t)(b * 9 + t) * CO + o) * CI) + i] = (_Float16)(wm[t] * f);
}

// ---------------------------------------------------------------------------
// Kernel 4: implicit-GEMM 3x3 stride-2 pad-1 conv on WMMA f16->f32.
// Per batch: M=256 (o), N=4096 (pixels), K=2304 (i x tap).
// Workgroup (8 waves): 128(o) x 128(px) output tile (2 output rows).
// Waves arranged 4(M) x 2(N): each wave owns a 32(o) x 64(px) subtile,
// i.e. 2 A fragments x 4 B fragments -> 8 WMMAs per K-step, halving
// B-fragment LDS traffic per WMMA vs a 16x128 wave tile.
// LDS stages the input patch [5 rows][130 cols][32 ch] per channel block;
// B fragments are single 32B contiguous LDS reads, A fragments come from
// global (weights are L2-resident).
// ---------------------------------------------------------------------------
__global__ __launch_bounds__(256) void k_conv(const _Float16* __restrict__ xb,
                                              const _Float16* __restrict__ weffL,
                                              const float* __restrict__ bias,
                                              float* __restrict__ out) {
  __shared__ __align__(32) _Float16 Bl[5][130][32];  // 41.6KB
  const int b = blockIdx.z;
  const int o_base = blockIdx.y * 128;
  const int p0 = blockIdx.x * 128;   // covers output rows y0, y0+1 (full width)
  const int y0 = p0 >> 6;
  const int tid = threadIdx.x;
  const int lane = tid & 31;
  const int wid = tid >> 5;          // 8 waves
  const int wm = wid & 3;            // M block: rows [32*wm, 32*wm+32)
  const int wn = wid >> 2;           // N block: pixels [64*wn, 64*wn+64)

  const _Float16* xbase = xb + (size_t)b * CI * HH * WW;

  v8f zero = {0.f, 0.f, 0.f, 0.f, 0.f, 0.f, 0.f, 0.f};
  v8f acc[2][4];
#pragma unroll
  for (int a = 0; a < 2; ++a)
#pragma unroll
    for (int nt = 0; nt < 4; ++nt) acc[a][nt] = zero;

  // A fragment mapping (16-bit A 16x32): lane L -> m = L%16;
  // half h -> k = h + 16*(L/16)  => one contiguous 16-half run.
  const int o_rowA0 = o_base + 32 * wm + (lane & 15);
  const int kbase = (lane >> 4) << 4;   // 0 or 16
  // B fragment mapping (32x16): lane L -> n = L%16; half h -> k = h + 16*(L/16)
  const int ccb = (lane >> 4) << 4;     // 0 or 16
  const int n = lane & 15;

  for (int ib = 0; ib < 8; ++ib) {      // channel blocks of 32
    const int i0 = ib * 32;
    __syncthreads();
    // Stage: Bl[r][c][cc] = x[b, i0+cc, 2*y0-1+r, c-1]  (zero padded)
    for (int idx = tid; idx < 5 * 130 * 32; idx += 256) {
      int cc = idx & 31;
      int rc = idx >> 5;
      int c = rc % 130, r = rc / 130;
      int giy = 2 * y0 - 1 + r;
      int gix = c - 1;
      _Float16 v = (_Float16)0.0f;
      if ((unsigned)giy < (unsigned)HH && (unsigned)gix < (unsigned)WW)
        v = xbase[((size_t)(i0 + cc) * HH + giy) * WW + gix];
      Bl[r][c][cc] = v;
    }
    __syncthreads();
    for (int ky = 0; ky < 3; ++ky) {
      for (int kx = 0; kx < 3; ++kx) {
        const int t = ky * 3 + kx;
        // Two A fragments: weffL[b][t][o_rowA0 + 16*a][i0+k]
        v16h afrag[2];
#pragma unroll
        for (int a = 0; a < 2; ++a) {
          const _Float16* wp = weffL +
              (((size_t)(b * 9 + t) * CO + (o_rowA0 + 16 * a)) * CI) + i0 + kbase;
          v8h a0 = *(const v8h*)(wp);
          v8h a1 = *(const v8h*)(wp + 8);
#pragma unroll
          for (int h = 0; h < 8; ++h) {
            afrag[a][h] = a0[h];
            afrag[a][h + 8] = a1[h];
          }
        }
#pragma unroll
        for (int nt = 0; nt < 4; ++nt) {
          const int pl = 64 * wn + 16 * nt + n;  // pixel within 128-px tile
          const int oyl = pl >> 6;               // 0 or 1
          const int ox = pl & 63;
          const int r = 2 * oyl + ky;            // staged row
          const int c = 2 * ox + kx;             // staged col (gix+1)
          v16h bfrag = *(const v16h*)&Bl[r][c][ccb];
#pragma unroll
          for (int a = 0; a < 2; ++a) {
            acc[a][nt] = __builtin_amdgcn_wmma_f32_16x16x32_f16(
                false, afrag[a], false, bfrag, (short)0, acc[a][nt], false,
                false);
          }
        }
      }
    }
  }
  // Epilogue: bias + scaled LeakyReLU, write NCHW f32.
  // C/D mapping: lane L -> n = L%16; VGPR r -> m = r + 8*(L/16).
  const float sqrt2 = 1.41421356237f;
#pragma unroll
  for (int nt = 0; nt < 4; ++nt) {
    const int pl = 64 * wn + 16 * nt + n;
    const int oy = y0 + (pl >> 6);
    const int ox = pl & 63;
#pragma unroll
    for (int a = 0; a < 2; ++a) {
#pragma unroll
      for (int r2 = 0; r2 < 8; ++r2) {
        const int o = o_base + 32 * wm + 16 * a + r2 + ((lane >> 4) << 3);
        float v = acc[a][nt][r2] + bias[o];
        v = (v >= 0.0f ? v : 0.2f * v) * sqrt2;
        out[(((size_t)b * CO + o) * HO + oy) * WO + ox] = v;
      }
    }
  }
}

// ---------------------------------------------------------------------------
extern "C" void kernel_launch(void* const* d_in, const int* in_sizes, int n_in,
                              void* d_out, int out_size, void* d_ws,
                              size_t ws_size, hipStream_t stream) {
  const float* x       = (const float*)d_in[0];  // [8,256,128,128]
  const float* s       = (const float*)d_in[1];  // [8,512]
  const float* conv_w  = (const float*)d_in[2];  // [256,256,3,3]
  const float* conv_b  = (const float*)d_in[3];  // [256]
  const float* style_w = (const float*)d_in[4];  // [256,512]
  const float* style_b = (const float*)d_in[5];  // [256]
  const float* norm_g  = (const float*)d_in[6];  // [256]
  float* out = (float*)d_out;                    // [8,256,64,64]

  char* ws = (char*)d_ws;
  size_t off = 0;
  auto take = [&](size_t bytes) {
    char* p = ws + off;
    off = (off + bytes + 255) & ~(size_t)255;
    return p;
  };
  _Float16* xb    = (_Float16*)take((size_t)BB * CI * HH * WW * 2);  // 64MB
  float*    sumsq = (float*)take((size_t)BB * CI * 4);
  float*    style = (float*)take((size_t)BB * CI * 4);
  _Float16* weffL = (_Float16*)take((size_t)BB * 9 * CO * CI * 2);   // 9MB
  (void)ws_size;

  hipMemsetAsync(sumsq, 0, (size_t)BB * CI * 4, stream);

  k_blur<<<BB * CI, 256, 0, stream>>>(x, xb, sumsq);
  k_style<<<(BB * CI + 255) / 256, 256, 0, stream>>>(s, style_w, style_b, style);
  k_wprep<<<BB * CO, 256, 0, stream>>>(conv_w, style, sumsq, norm_g, weffL);
  dim3 grid(32, 2, BB);  // 32 pixel-blocks x 2 o-blocks x 8 batches
  k_conv<<<grid, 256, 0, stream>>>(xb, weffL, conv_b, out);
}
